// PE_86663850098730
// MI455X (gfx1250) — matrix operand order repaired and checked by reading, not compiled, not run on hardware
//
#include <hip/hip_runtime.h>
#include <hip/hip_bf16.h>

// ---------------------------------------------------------------------------
// Point-cloud upsampling pipeline for MI455X (gfx1250, wave32, WMMA).
// Stages: knn -> mlp1(fp16 out) -> cov+mlp2 -> WMMA attention (fused
// gemm1/relu/gemm2/softmax/aggregate) -> mlp3 + output assembly.
// All intermediates fit in the 192MB L2; compute-bound on the attention
// GEMMs which run on v_wmma_f32_16x16x32_f16.
// F/H tiles stored column-major with padded strides so B-fragment slot
// indices (K = kc*32 + 16g + s, contiguous in s) map to contiguous LDS ->
// ds_load_b128 instead of 16x ds_load_u16 + v_mov_b16 packing.
// ---------------------------------------------------------------------------

typedef __attribute__((ext_vector_type(16))) _Float16 v16h;
typedef __attribute__((ext_vector_type(8)))  float    v8f;

#define NPTS 8192
#define NB   2
#define KNN  16

// ------------------------------- kNN ---------------------------------------
// One thread per query point; candidates staged through LDS in 1024-point
// tiles; top-16 kept as a sorted register array (single bubble pass per
// accepted candidate). Next tile prefetched with global_prefetch_b8.
#define KNN_CH 1024
__global__ __launch_bounds__(256) void knn_kernel(const float* __restrict__ x,
                                                  int* __restrict__ idxb) {
  __shared__ float px[KNN_CH], py[KNN_CH], pz[KNN_CH];
  const int N = NPTS;
  int b = blockIdx.y;
  int n = blockIdx.x * 256 + threadIdx.x;
  const float* xb = x + b * 3 * N;
  float qx = xb[n], qy = xb[N + n], qz = xb[2 * N + n];
  float bd[KNN]; int bi[KNN];
#pragma unroll
  for (int k = 0; k < KNN; ++k) { bd[k] = 3.4e38f; bi[k] = n; }
  for (int ch = 0; ch < N / KNN_CH; ++ch) {
    __syncthreads();
    for (int t = threadIdx.x; t < KNN_CH; t += 256) {
      int gi = ch * KNN_CH + t;
      px[t] = xb[gi]; py[t] = xb[N + gi]; pz[t] = xb[2 * N + gi];
      if (ch + 1 < N / KNN_CH) {                 // stream next tile into caches
        __builtin_prefetch(&xb[gi + KNN_CH], 0, 0);
        __builtin_prefetch(&xb[N + gi + KNN_CH], 0, 0);
        __builtin_prefetch(&xb[2 * N + gi + KNN_CH], 0, 0);
      }
    }
    __syncthreads();
    for (int t = 0; t < KNN_CH; ++t) {
      float dx = px[t] - qx, dy = py[t] - qy, dz = pz[t] - qz;
      float d = dx * dx + dy * dy + dz * dz;
      if (d < bd[KNN - 1]) {
        bd[KNN - 1] = d; bi[KNN - 1] = ch * KNN_CH + t;
#pragma unroll
        for (int k = KNN - 1; k > 0; --k) {      // one bubble pass (sorted)
          if (bd[k] < bd[k - 1]) {
            float td = bd[k]; bd[k] = bd[k - 1]; bd[k - 1] = td;
            int   ti = bi[k]; bi[k] = bi[k - 1]; bi[k - 1] = ti;
          }
        }
      }
    }
  }
#pragma unroll
  for (int k = 0; k < KNN; ++k) idxb[(b * N + n) * KNN + k] = bi[k];
}

// ------------------------------ mlp1 ---------------------------------------
// Pointwise 3->32->64, ReLU both layers; result stored fp16 per point
// ([pt][64] contiguous) as WMMA feed for the attention stage.
__global__ __launch_bounds__(256) void mlp1_kernel(const float* __restrict__ x,
                                                   const float* __restrict__ w1,
                                                   const float* __restrict__ b1,
                                                   const float* __restrict__ w2,
                                                   const float* __restrict__ b2,
                                                   _Float16* __restrict__ f1h) {
  __shared__ float w1s[32 * 3], b1s[32], w2s[64 * 32], b2s[64];
  int tid = threadIdx.x;
  for (int t = tid; t < 96;   t += 256) w1s[t] = w1[t];
  for (int t = tid; t < 32;   t += 256) b1s[t] = b1[t];
  for (int t = tid; t < 2048; t += 256) w2s[t] = w2[t];
  for (int t = tid; t < 64;   t += 256) b2s[t] = b2[t];
  __syncthreads();
  const int N = NPTS;
  int pt = blockIdx.x * 256 + tid;
  int b = pt >> 13, n = pt & (N - 1);
  const float* xb = x + b * 3 * N;
  float x0 = xb[n], x1 = xb[N + n], x2 = xb[2 * N + n];
  float h[32];
#pragma unroll
  for (int o = 0; o < 32; ++o) {
    float a = w1s[o * 3] * x0 + w1s[o * 3 + 1] * x1 + w1s[o * 3 + 2] * x2 + b1s[o];
    h[o] = a > 0.f ? a : 0.f;
  }
  for (int o = 0; o < 64; ++o) {
    float a = b2s[o];
#pragma unroll
    for (int i = 0; i < 32; ++i) a += w2s[o * 32 + i] * h[i];
    a = a > 0.f ? a : 0.f;
    f1h[pt * 64 + o] = (_Float16)a;
  }
}

// --------------------------- cov + mlp2 ------------------------------------
// Single-pass raw-moment covariance (E[pp^T] - m m^T == mean-centered cov),
// then 9->32 ReLU MLP into featbuf[pt][64..95].
__global__ __launch_bounds__(256) void cov_mlp2_kernel(const float* __restrict__ x,
                                                       const int* __restrict__ idxb,
                                                       const float* __restrict__ w,
                                                       const float* __restrict__ bias,
                                                       float* __restrict__ featbuf) {
  __shared__ float ws[32 * 9], bs[32];
  int tid = threadIdx.x;
  for (int t = tid; t < 288; t += 256) ws[t] = w[t];
  if (tid < 32) bs[tid] = bias[tid];
  __syncthreads();
  const int N = NPTS;
  int pt = blockIdx.x * 256 + tid;
  int b = pt >> 13;
  const float* xb = x + b * 3 * N;
  float sx = 0, sy = 0, sz = 0, sxx = 0, sxy = 0, sxz = 0, syy = 0, syz = 0, szz = 0;
#pragma unroll
  for (int k = 0; k < KNN; ++k) {
    int id = idxb[pt * KNN + k];
    float cx = xb[id], cy = xb[N + id], cz = xb[2 * N + id];
    sx += cx; sy += cy; sz += cz;
    sxx += cx * cx; sxy += cx * cy; sxz += cx * cz;
    syy += cy * cy; syz += cy * cz; szz += cz * cz;
  }
  const float inv = 1.f / (float)KNN;
  float mx = sx * inv, my = sy * inv, mz = sz * inv;
  float c9[9];
  c9[0] = sxx * inv - mx * mx; c9[1] = sxy * inv - mx * my; c9[2] = sxz * inv - mx * mz;
  c9[3] = c9[1];               c9[4] = syy * inv - my * my; c9[5] = syz * inv - my * mz;
  c9[6] = c9[2];               c9[7] = c9[5];               c9[8] = szz * inv - mz * mz;
  for (int o = 0; o < 32; ++o) {
    float a = bs[o];
#pragma unroll
    for (int i = 0; i < 9; ++i) a += ws[o * 9 + i] * c9[i];
    featbuf[pt * 96 + 64 + o] = a > 0.f ? a : 0.f;
  }
}

// -------------------- WMMA attention (the hot kernel) ----------------------
// One wave handles one point's 16-column tile (col 0 = self, relative feature
// exactly 0, masked out of the softmax). GEMM1 (128x64 @ 64x16) and GEMM2
// (64x128 @ 128x16) via v_wmma_f32_16x16x32_f16, H staged per-wave in LDS.
// F/H are column-major with padded strides: B-fragment loads and producer
// stores are contiguous 16B chunks (ds_load_b128/ds_store_b128), and the
// padded strides (36 resp. 68 dwords) spread the 16 columns across distinct
// LDS banks. Softmax + aggregation via 16-lane shfl_xor trees.
// att_b2 is omitted: a per-channel logit shift is softmax-invariant.
#define AT_WAVES 4
#define FSTR 72    // halves per F column (64 + 8 pad) -> 144B, 36-dword stride
#define HSTR 136   // halves per H column (128 + 8 pad) -> 272B, 68-dword stride
__global__ __launch_bounds__(128) void attn_kernel(const float* __restrict__ att_w1,
                                                   const float* __restrict__ att_b1,
                                                   const float* __restrict__ att_w2,
                                                   const _Float16* __restrict__ f1h,
                                                   const int* __restrict__ idxb,
                                                   float* __restrict__ featbuf) {
  __shared__ _Float16 W1h[128 * 64];              // 16 KB   [out][in]
  __shared__ _Float16 W2h[64 * 128];              // 16 KB   [out][in]
  __shared__ float    b1s[128];                   // 0.5 KB
  __shared__ _Float16 Fs[AT_WAVES][16 * FSTR];    // 9 KB    [col][k]
  __shared__ _Float16 Hs[AT_WAVES][16 * HSTR];    // 17 KB   [col][k]

  int tid = threadIdx.x;
  for (int t = tid; t < 128 * 64; t += 128) W1h[t] = (_Float16)att_w1[t];
  for (int t = tid; t < 64 * 128; t += 128) W2h[t] = (_Float16)att_w2[t];
  if (tid < 128) b1s[tid] = att_b1[tid];
  __syncthreads();

  int wave = tid >> 5;
  int lane = tid & 31;
  int g    = lane >> 4;     // K-half select (CDNA5 wave32 fragment layout)
  int mn   = lane & 15;     // A: row; B/C/D: column
  _Float16* F = &Fs[wave][0];
  _Float16* H = &Hs[wave][0];

  int gwave = blockIdx.x * AT_WAVES + wave;
  for (int it = 0; it < 8; ++it) {
    int pt = gwave * 8 + it;                      // 0 .. B*N-1
    int b = pt >> 13;
    int nbase = b << 13;
    int col = mn;
    int nbr = idxb[pt * KNN + col];               // col 0 == self -> diff 0

    // Build F (column-major, [col][k]): 32 contiguous halves per lane.
    const _Float16* fsel = f1h + pt * 64 + g * 32;
    const _Float16* fngb = f1h + (nbase + nbr) * 64 + g * 32;
#pragma unroll
    for (int c = 0; c < 32; ++c)
      F[col * FSTR + g * 32 + c] = (_Float16)(fngb[c] - fsel[c]);
    asm volatile("s_wait_dscnt 0" ::: "memory");  // wave-local LDS RAW

    // B fragments for GEMM1 (K=64 as 2 x k32): contiguous 16 halves/lane.
    v16h bf[2];
#pragma unroll
    for (int kc = 0; kc < 2; ++kc)
#pragma unroll
      for (int s = 0; s < 16; ++s)
        bf[kc][s] = F[col * FSTR + kc * 32 + 16 * g + s];

    // GEMM1: H = relu(W1 @ F + b1), 8 row tiles.
#pragma unroll
    for (int m = 0; m < 8; ++m) {
      v8f acc = {};
#pragma unroll
      for (int kc = 0; kc < 2; ++kc) {
        v16h a;
#pragma unroll
        for (int s = 0; s < 16; ++s) {
          int ka = kc * 32 + 16 * g + s; // A layout
          a[s] = W1h[(16 * m + mn) * 64 + ka];
        }
        acc = __builtin_amdgcn_wmma_f32_16x16x32_f16(false, a, false, bf[kc],
                                                     (short)0, acc, false, false);
      }
      // Writeback: rows 16m+8g+vi contiguous -> one 16B store per lane.
#pragma unroll
      for (int vi = 0; vi < 8; ++vi) {
        int row = 16 * m + 8 * g + vi;            // C/D layout
        float v = acc[vi] + b1s[row];
        v = v > 0.f ? v : 0.f;
        H[mn * HSTR + row] = (_Float16)v;
      }
    }
    asm volatile("s_wait_dscnt 0" ::: "memory");

    // B fragments for GEMM2 (K=128 as 4 x k32): contiguous 16 halves/lane.
    v16h hf[4];
#pragma unroll
    for (int kc = 0; kc < 4; ++kc)
#pragma unroll
      for (int s = 0; s < 16; ++s)
        hf[kc][s] = H[mn * HSTR + kc * 32 + 16 * g + s];

    // GEMM2: logits = W2 @ H (bias dropped: softmax-invariant), then
    // per-row softmax over cols 1..15 and aggregation of F.
#pragma unroll
    for (int r = 0; r < 4; ++r) {
      v8f acc = {};
#pragma unroll
      for (int kc = 0; kc < 4; ++kc) {
        v16h a;
#pragma unroll
        for (int s = 0; s < 16; ++s) {
          int ka = kc * 32 + 16 * g + s;
          a[s] = W2h[(16 * r + mn) * 128 + ka];
        }
        acc = __builtin_amdgcn_wmma_f32_16x16x32_f16(false, a, false, hf[kc],
                                                     (short)0, acc, false, false);
      }
#pragma unroll
      for (int vi = 0; vi < 8; ++vi) {
        int row = 16 * r + 8 * g + vi;
        float v = (col == 0) ? -3.0e38f : acc[vi];
        float m1 = v;                              // row max over 16-lane group
        m1 = fmaxf(m1, __shfl_xor(m1, 1));
        m1 = fmaxf(m1, __shfl_xor(m1, 2));
        m1 = fmaxf(m1, __shfl_xor(m1, 4));
        m1 = fmaxf(m1, __shfl_xor(m1, 8));
        float e = (col == 0) ? 0.f : __expf(v - m1);
        float es = e;
        es += __shfl_xor(es, 1);
        es += __shfl_xor(es, 2);
        es += __shfl_xor(es, 4);
        es += __shfl_xor(es, 8);
        float wgt = e / es;
        float part = wgt * (float)F[col * FSTR + row];
        part += __shfl_xor(part, 1);
        part += __shfl_xor(part, 2);
        part += __shfl_xor(part, 4);
        part += __shfl_xor(part, 8);
        if (col == 1) featbuf[pt * 96 + row] = part;   // f_agg
      }
    }
  }
}

// ---------------------- mlp3 (96->128->12) + output ------------------------
// Block of 128 threads handles 32 points; feat & hidden tiles in LDS; weight
// indices are wave-uniform -> scalar (SMEM) loads from L2-resident weights.
__global__ __launch_bounds__(128) void mlp3_out_kernel(const float* __restrict__ x,
                                                       const float* __restrict__ w31,
                                                       const float* __restrict__ b31,
                                                       const float* __restrict__ w32,
                                                       const float* __restrict__ b32,
                                                       const float* __restrict__ featbuf,
                                                       float* __restrict__ out) {
  __shared__ float fs[32][96];    // 12 KB
  __shared__ float hs[32][128];   // 16 KB
  int tid = threadIdx.x;
  int p0 = blockIdx.x * 32;
  for (int t = tid; t < 32 * 96; t += 128)
    fs[t / 96][t % 96] = featbuf[(p0 + t / 96) * 96 + (t % 96)];
  __syncthreads();
  int p = tid & 31;
  int grp = tid >> 5;                             // uniform within a wave
  for (int oo = 0; oo < 32; ++oo) {
    int o = grp + oo * 4;
    float a = b31[o];
    for (int i = 0; i < 96; ++i) a += w31[o * 96 + i] * fs[p][i];
    hs[p][o] = a > 0.f ? a : 0.f;
  }
  __syncthreads();
  const int N = NPTS;
  int pt = p0 + p;
  int b = pt >> 13, n = pt & (N - 1);
  for (int cc = 0; cc < 3; ++cc) {
    int c = grp * 3 + cc;                         // 0..11
    float a = b32[c];
    for (int o = 0; o < 128; ++o) a += w32[c * 128 + o] * hs[p][o];
    int i = c >> 2, j = c & 3;                    // g(B,12,N) -> (B,3,4N)
    out[b * 3 * 4 * N + i * 4 * N + j * N + n] =
        x[b * 3 * N + i * N + n] + 0.15f * a;
  }
}

// ------------------------------ launcher -----------------------------------
extern "C" void kernel_launch(void* const* d_in, const int* in_sizes, int n_in,
                              void* d_out, int out_size, void* d_ws, size_t ws_size,
                              hipStream_t stream) {
  const float* x     = (const float*)d_in[0];
  const float* m1w1  = (const float*)d_in[1];
  const float* m1b1  = (const float*)d_in[2];
  const float* m1w2  = (const float*)d_in[3];
  const float* m1b2  = (const float*)d_in[4];
  const float* m2w1  = (const float*)d_in[5];
  const float* m2b1  = (const float*)d_in[6];
  const float* m3w1  = (const float*)d_in[7];
  const float* m3b1  = (const float*)d_in[8];
  const float* m3w2  = (const float*)d_in[9];
  const float* m3b2  = (const float*)d_in[10];
  const float* aw1   = (const float*)d_in[11];
  const float* ab1   = (const float*)d_in[12];
  const float* aw2   = (const float*)d_in[13];
  // att_b2 (d_in[14]) unused: per-channel logit bias is softmax-invariant.

  char* ws = (char*)d_ws;
  int*      idxb    = (int*)ws;                           // 1 MB
  _Float16* f1h     = (_Float16*)(ws + (1u << 20));       // 2 MB
  float*    featbuf = (float*)(ws + 3u * (1u << 20));     // 6 MB
  float*    outp    = (float*)d_out;

  knn_kernel     <<<dim3(NPTS / 256, NB), 256, 0, stream>>>(x, idxb);
  mlp1_kernel    <<<(NB * NPTS) / 256, 256, 0, stream>>>(x, m1w1, m1b1, m1w2, m1b2, f1h);
  cov_mlp2_kernel<<<(NB * NPTS) / 256, 256, 0, stream>>>(x, idxb, m2w1, m2b1, featbuf);
  attn_kernel    <<<(NB * NPTS) / (AT_WAVES * 8), 128, 0, stream>>>(aw1, ab1, aw2, f1h, idxb, featbuf);
  mlp3_out_kernel<<<(NB * NPTS) / 32, 128, 0, stream>>>(x, m3w1, m3b1, m3w2, m3b2, featbuf, outp);
}
